// AttentiveRNNClassifier_13340168421398
// MI455X (gfx1250) — matrix operand order repaired and checked by reading, not compiled, not run on hardware
//
#include <hip/hip_runtime.h>
#include <hip/hip_bf16.h>
#include <math.h>

typedef __attribute__((ext_vector_type(16))) _Float16 v16h;
typedef __attribute__((ext_vector_type(8)))  _Float16 v8h;
typedef __attribute__((ext_vector_type(4)))  _Float16 v4h;
typedef __attribute__((ext_vector_type(8)))  float    v8f;

#define EMB   256
#define HID   256
#define LABEL 5
#define BB    64
#define SS    512
#define G3    768          // 3*HID
#define NROW  32768        // B*S
#define LDSW  264          // padded LDS row (halves) -> conflict-free ds_load_b128
#define GXSL  (BB * 384)   // per-step gate-input slice (halves) for one WG half

// ---------------- WMMA helpers (CDNA5 v_wmma_f32_16x16x32_f16) ----------------

__device__ __forceinline__ v8f zero8() {
  v8f z;
#pragma unroll
  for (int i = 0; i < 8; ++i) z[i] = 0.f;
  return z;
}

__device__ __forceinline__ v8f wmma16(v16h a, v16h b, v8f c) {
  return __builtin_amdgcn_wmma_f32_16x16x32_f16(false, a, false, b, (short)0, c, false, false);
}

// A fragment (16x32 f16, MxK) from row-major [rows x ld], tile at (row0, k0).
// Lane l<16: row=row0+l, halves K = k0+{0..15}
// Lane l>=16: row=row0+l-16, halves K = k0+{16..31}
__device__ __forceinline__ v16h load_a_frag(const _Float16* __restrict__ m, int ld,
                                            int row0, int k0, int lane) {
  int lr = lane & 15, hi = lane >> 4;
  const _Float16* p = m + (size_t)(row0 + lr) * ld + k0 + hi * 16;
  v8h c0 = *(const v8h*)p;
  v8h c1 = *(const v8h*)(p + 8);
  v16h a;
#pragma unroll
  for (int i = 0; i < 8; ++i) { a[i] = c0[i]; a[i + 8] = c1[i]; }
  return a;
}

// B fragment (32x16 f16, KxN) where B[k][n] = W[col0+n][k0+k] (W row-major, ld = K stride).
// Lane l: col n = l&15, halves K = k0 + (l>=16?16:0) + {0..15} (contiguous 32B run).
__device__ __forceinline__ v16h load_b_frag(const _Float16* __restrict__ w, int ld,
                                            int col0, int k0, int lane) {
  int n = lane & 15, hi = lane >> 4;
  const _Float16* p = w + (size_t)(col0 + n) * ld + k0 + hi * 16;
  v8h c0 = *(const v8h*)p;
  v8h c1 = *(const v8h*)(p + 8);
  v16h b;
#pragma unroll
  for (int i = 0; i < 8; ++i) { b[i] = c0[i]; b[i + 8] = c1[i]; }
  return b;
}

// Same B fragment from LDS (padded row LDSW halves) -> ds_load_b128 x2, conflict-free
__device__ __forceinline__ v16h load_b_lds(const _Float16* sW, int row0, int kt, int lane) {
  int n = lane & 15, hi = lane >> 4;
  const _Float16* p = sW + (row0 + n) * LDSW + kt * 32 + hi * 16;
  v8h c0 = *(const v8h*)p;
  v8h c1 = *(const v8h*)(p + 8);
  v16h b;
#pragma unroll
  for (int i = 0; i < 8; ++i) { b[i] = c0[i]; b[i + 8] = c1[i]; }
  return b;
}

// ---------------- CDNA5 async global->LDS copy (ASYNCcnt-tracked) ----------------

__device__ __forceinline__ void async_g2l_b128(const _Float16* g, const _Float16* l) {
  unsigned ldsOff = (unsigned)(uintptr_t)l;   // flat LDS aperture truncates to LDS offset
  asm volatile("global_load_async_to_lds_b128 %0, %1, off"
               :: "v"(ldsOff), "v"(g)
               : "memory");
}

__device__ __forceinline__ void wait_async0() {
  asm volatile("s_wait_asynccnt 0x0" ::: "memory");
}

// ---------------- Phase 0: init (score mask/bias, sync counters) ----------------

__global__ void __launch_bounds__(256) init_kernel(float* __restrict__ score,
                                                   const int* __restrict__ len,
                                                   const float* __restrict__ ba,
                                                   unsigned* __restrict__ ctr) {
  int i = blockIdx.x * 256 + threadIdx.x;
  if (i < 8) ctr[i] = 0u;
  if (i < NROW) {
    int b = i >> 9, s = i & 511;
    score[i] = ba[0] + ((s >= len[b]) ? -10000.f : 0.f);
  }
}

// ---------------- Phase 1: embedding gather -> f16, weight converts ----------------

__global__ void __launch_bounds__(256) embed_kernel(const int* __restrict__ seq,
                                                    const float* __restrict__ emb,
                                                    _Float16* __restrict__ x16) {
  int i = blockIdx.x * 256 + threadIdx.x;       // NROW*64 threads, 4 elems each
  int row = i >> 6;
  int e0 = (i & 63) * 4;
  int tok = seq[row];
  float4 v = *(const float4*)(emb + (size_t)tok * EMB + e0);
  v4h h;
  h[0] = (_Float16)v.x; h[1] = (_Float16)v.y; h[2] = (_Float16)v.z; h[3] = (_Float16)v.w;
  *(v4h*)(x16 + (size_t)row * EMB + e0) = h;
}

__global__ void __launch_bounds__(256) cvt_kernel(const float* __restrict__ src,
                                                  _Float16* __restrict__ dst, int n4) {
  int i = blockIdx.x * 256 + threadIdx.x;
  if (i < n4) {
    float4 v = *(const float4*)(src + (size_t)i * 4);
    v4h h;
    h[0] = (_Float16)v.x; h[1] = (_Float16)v.y; h[2] = (_Float16)v.z; h[3] = (_Float16)v.w;
    *(v4h*)(dst + (size_t)i * 4) = h;
  }
}

// ---------------- Phase 2: gx = x @ Wih^T + bih (both directions), stored f16 ----------------
// waves: 2 dirs x 2048 mtiles x 12 ngroups (4 ntiles each) = 49152 -> 6144 blocks

__global__ void __launch_bounds__(256) gx_gemm_kernel(const _Float16* __restrict__ x16,
                                                      const _Float16* __restrict__ wih16,
                                                      const float* __restrict__ bih_f,
                                                      const float* __restrict__ bih_b,
                                                      _Float16* __restrict__ gx16) {
  int gw = blockIdx.x * 8 + (threadIdx.x >> 5);
  int lane = threadIdx.x & 31;
  int lr = lane & 15, hi = lane >> 4;
  int dir = gw / 24576;
  int rem = gw % 24576;
  int mt = rem / 12, ng = rem % 12;
  const _Float16* W = wih16 + (size_t)dir * (G3 * EMB);
  const float* bih = dir ? bih_b : bih_f;
  _Float16* gx = gx16 + (size_t)dir * ((size_t)NROW * G3);

  v8f acc[4];
#pragma unroll
  for (int nj = 0; nj < 4; ++nj) acc[nj] = zero8();

  for (int kt = 0; kt < 8; ++kt) {
    v16h a = load_a_frag(x16, EMB, mt * 16, kt * 32, lane);
#pragma unroll
    for (int nj = 0; nj < 4; ++nj) {
      v16h b = load_b_frag(W, EMB, (ng * 4 + nj) * 16, kt * 32, lane);
      acc[nj] = wmma16(a, b, acc[nj]);
    }
  }
#pragma unroll
  for (int nj = 0; nj < 4; ++nj) {
    int col = (ng * 4 + nj) * 16 + lr;
    float bias = bih[col];
#pragma unroll
    for (int v = 0; v < 8; ++v) {
      int r = mt * 16 + v + hi * 8;
      gx[(size_t)r * G3 + col] = (_Float16)(acc[nj][v] + bias);
    }
  }
}

// ---------------- Phase 3: persistent bidirectional GRU scan ----------------
// 4 blocks: (dir, half). Each block owns j in [half*128, half*128+128), keeps the
// matching 384x256 slice of Whh in LDS (f16, padded rows, 198 KB), and double-buffers
// each step's gate-input slice (64x384 halves, 48 KB x2) in LDS via
// GLOBAL_LOAD_ASYNC_TO_LDS_B128 issued one step ahead (ASYNCcnt pipelined against
// the WMMA work and the inter-WG barrier). Gates r/z/n for the owned j-half are
// computed entirely in WMMA accumulators; h_prev stays in registers; the 64x256 h
// vector is exchanged through a double-buffered L2-resident global buffer with one
// release/acquire barrier per step.

__global__ void __launch_bounds__(256) gru_scan_kernel(const _Float16* __restrict__ gx16,
                                                       const _Float16* __restrict__ whh16,
                                                       const float* __restrict__ bhh_f,
                                                       const float* __restrict__ bhh_b,
                                                       _Float16* __restrict__ hbuf,
                                                       float* __restrict__ rnn32,
                                                       _Float16* __restrict__ rnn16,
                                                       unsigned* __restrict__ ctr) {
  extern __shared__ _Float16 smem[];
  _Float16* sW  = smem;                 // [384][LDSW]   Whh slice fragments
  _Float16* sGX = smem + 384 * LDSW;    // [2][64][384]  double-buffered gate inputs

  const int dir = blockIdx.x >> 1;
  const int wg  = blockIdx.x & 1;
  const int tid = threadIdx.x;
  const int lane = tid & 31, wv = tid >> 5;
  const int lr = lane & 15, hi = lane >> 4;

  const _Float16* gx  = gx16  + (size_t)dir * ((size_t)NROW * G3);
  const _Float16* Whh = whh16 + (size_t)dir * (G3 * HID);
  const float*    bhh = dir ? bhh_b : bhh_f;
  _Float16*       hb  = hbuf + (size_t)dir * (2 * BB * HID);

  // LDS fill: slice rows q: gate=q/128, jj=q%128 -> global row gate*256 + wg*128 + jj
  for (int chunk = tid; chunk < 384 * 32; chunk += 256) {
    int q = chunk >> 5;
    int k = (chunk & 31) * 8;
    int g = (q >> 7) * 256 + wg * 128 + (q & 127);
    *(v8h*)(sW + q * LDSW + k) = *(const v8h*)(Whh + (size_t)g * HID + k);
  }
  // zero h0 (buffer 0, own j-half)
  for (int i = tid; i < BB * 16; i += 256) {
    int b = i >> 4;
    int j0 = (i & 15) * 8;
    v8h z = {};
    *(v8h*)(hb + b * HID + wg * 128 + j0) = z;
  }

  // async stage of step-t gate inputs: 3072 b128 chunks -> 12 per thread
  // chunk c: b = c/48, gate = (c%48)/16, q = c%16 (8 halves each)
  auto stage_gx = [&](int s, int buf) {
#pragma unroll
    for (int i = 0; i < 12; ++i) {
      int c = tid + i * 256;
      int b = c / 48;
      int r = c % 48;
      int gate = r >> 4;
      int q = r & 15;
      const _Float16* src = gx + ((size_t)b * SS + s) * G3 + gate * 256 + wg * 128 + q * 8;
      const _Float16* dst = sGX + buf * GXSL + b * 384 + gate * 128 + q * 8;
      async_g2l_b128(src, dst);
    }
  };

  const int mtile  = wv & 3;
  const int jgbase = (wv >> 2) * 4;

  float br[4], bz[4], bn[4];
#pragma unroll
  for (int u = 0; u < 4; ++u) {
    int j = wg * 128 + (jgbase + u) * 16 + lr;
    br[u] = bhh[j]; bz[u] = bhh[256 + j]; bn[u] = bhh[512 + j];
  }

  v8f hprev[4];
#pragma unroll
  for (int u = 0; u < 4; ++u) hprev[u] = zero8();

  int phase = 0;
  auto gsync = [&]() {
    __threadfence();
    __syncthreads();
    ++phase;
    if (tid == 0) {
      __hip_atomic_fetch_add(&ctr[dir], 1u, __ATOMIC_RELEASE, __HIP_MEMORY_SCOPE_AGENT);
      unsigned tgt = 2u * (unsigned)phase;
      while (__hip_atomic_load(&ctr[dir], __ATOMIC_ACQUIRE, __HIP_MEMORY_SCOPE_AGENT) < tgt)
        __builtin_amdgcn_s_sleep(1);
    }
    __syncthreads();
  };

  // prologue: start async copy of step-0 gate inputs, then make sW + h0 visible
  stage_gx(dir ? (SS - 1) : 0, 0);
  gsync();

  for (int t = 0; t < SS; ++t) {
    const int s = dir ? (SS - 1 - t) : t;
    const _Float16* hin  = hb + (t & 1) * (BB * HID);
    _Float16*       hout = hb + ((t + 1) & 1) * (BB * HID);

    // gate inputs for step t have landed in LDS; publish block-wide
    wait_async0();
    __syncthreads();

    // kick off next step's gate-input copy (overlaps this step's WMMAs + barrier)
    if (t + 1 < SS) stage_gx(dir ? (s - 1) : (s + 1), (t + 1) & 1);

    const _Float16* gxl = sGX + (t & 1) * GXSL;

    v16h af[8];
#pragma unroll
    for (int kt = 0; kt < 8; ++kt)
      af[kt] = load_a_frag(hin, HID, mtile * 16, kt * 32, lane);

#pragma unroll
    for (int u = 0; u < 4; ++u) {
      const int jg = jgbase + u;
      v8f ar = zero8(), az = zero8(), an = zero8();
#pragma unroll
      for (int kt = 0; kt < 8; ++kt) {
        v16h b_r = load_b_lds(sW,       jg * 16, kt, lane);
        v16h b_z = load_b_lds(sW, 128 + jg * 16, kt, lane);
        v16h b_n = load_b_lds(sW, 256 + jg * 16, kt, lane);
        ar = wmma16(af[kt], b_r, ar);
        az = wmma16(af[kt], b_z, az);
        an = wmma16(af[kt], b_n, an);
      }
      const int jl = jg * 16 + lr;          // j within own 128-col half
      const int j  = wg * 128 + jl;         // global hidden index
#pragma unroll
      for (int v = 0; v < 8; ++v) {
        const int b = mtile * 16 + v + hi * 8;
        float xr = (float)gxl[b * 384 + jl];
        float xz = (float)gxl[b * 384 + 128 + jl];
        float xn = (float)gxl[b * 384 + 256 + jl];
        float r = 1.f / (1.f + __expf(-(xr + ar[v] + br[u])));
        float z = 1.f / (1.f + __expf(-(xz + az[v] + bz[u])));
        float n = tanhf(xn + r * (an[v] + bn[u]));
        float h = (1.f - z) * n + z * hprev[u][v];
        hprev[u][v] = h;
        const size_t o = ((size_t)b * SS + s) * (2 * HID) + dir * HID + j;
        rnn32[o] = h;
        rnn16[o] = (_Float16)h;
        hout[b * HID + j] = (_Float16)h;
      }
    }
    gsync();   // h(t+1) complete before anyone reads it
  }
}

// ---------------- Phase 4a: score[b,s] += Wa . tanh(rnn @ Ww^T + bw) ----------------
// waves: 2048 mtiles x 4 ngroups = 8192 -> 1024 blocks

__global__ void __launch_bounds__(256) attn_score_kernel(const _Float16* __restrict__ rnn16,
                                                         const _Float16* __restrict__ ww16,
                                                         const float* __restrict__ bw,
                                                         const float* __restrict__ wa,
                                                         float* __restrict__ score) {
  int gw = blockIdx.x * 8 + (threadIdx.x >> 5);
  int lane = threadIdx.x & 31;
  int lr = lane & 15, hi = lane >> 4;
  int mt = gw >> 2, ng = gw & 3;

  v8f acc[4];
#pragma unroll
  for (int nj = 0; nj < 4; ++nj) acc[nj] = zero8();

  for (int kt = 0; kt < 16; ++kt) {
    v16h a = load_a_frag(rnn16, 2 * HID, mt * 16, kt * 32, lane);
#pragma unroll
    for (int nj = 0; nj < 4; ++nj) {
      v16h b = load_b_frag(ww16, 2 * HID, (ng * 4 + nj) * 16, kt * 32, lane);
      acc[nj] = wmma16(a, b, acc[nj]);
    }
  }
#pragma unroll
  for (int nj = 0; nj < 4; ++nj) {
    int n = (ng * 4 + nj) * 16 + lr;
    float bwv = bw[n], wav = wa[n];
#pragma unroll
    for (int v = 0; v < 8; ++v) {
      float val = tanhf(acc[nj][v] + bwv) * wav;
      val += __shfl_xor(val, 1, 16);
      val += __shfl_xor(val, 2, 16);
      val += __shfl_xor(val, 4, 16);
      val += __shfl_xor(val, 8, 16);
      if (lr == 0) atomicAdd(&score[mt * 16 + v + hi * 8], val);
    }
  }
}

// ---------------- Phase 4b: softmax over S, encoding, classifier, output ----------------

__global__ void __launch_bounds__(256) finalize_kernel(const float* __restrict__ score,
                                                       const float* __restrict__ rnn32,
                                                       const float* __restrict__ Wc,
                                                       const float* __restrict__ bc,
                                                       float* __restrict__ out) {
  __shared__ float sAttn[SS];
  __shared__ float sRed[256];
  __shared__ float sEnc[2 * HID];
  __shared__ float sLog[LABEL];
  const int b = blockIdx.x, tid = threadIdx.x;

  float v0 = score[b * SS + tid];
  float v1 = score[b * SS + 256 + tid];
  sRed[tid] = fmaxf(v0, v1);
  __syncthreads();
  for (int o = 128; o > 0; o >>= 1) {
    if (tid < o) sRed[tid] = fmaxf(sRed[tid], sRed[tid + o]);
    __syncthreads();
  }
  float mx = sRed[0];
  __syncthreads();
  float e0 = __expf(v0 - mx), e1 = __expf(v1 - mx);
  sAttn[tid] = e0; sAttn[tid + 256] = e1;
  sRed[tid] = e0 + e1;
  __syncthreads();
  for (int o = 128; o > 0; o >>= 1) {
    if (tid < o) sRed[tid] += sRed[tid + o];
    __syncthreads();
  }
  float inv = 1.f / sRed[0];
  __syncthreads();
  sAttn[tid] *= inv; sAttn[tid + 256] *= inv;
  __syncthreads();

  float a0 = 0.f, a1 = 0.f;
  for (int s = 0; s < SS; ++s) {
    float w = sAttn[s];
    const float* row = rnn32 + ((size_t)b * SS + s) * (2 * HID);
    a0 += w * row[tid];
    a1 += w * row[tid + 256];
  }
  sEnc[tid] = a0; sEnc[tid + 256] = a1;
  __syncthreads();
  if (tid < LABEL) {
    float acc = bc[tid];
    const float* w = Wc + tid * (2 * HID);
    for (int k = 0; k < 2 * HID; ++k) acc += w[k] * sEnc[k];
    sLog[tid] = acc;
  }
  __syncthreads();
  if (tid == 0) {
    float m2 = sLog[0];
    for (int o = 1; o < LABEL; ++o) m2 = fmaxf(m2, sLog[o]);
    float ex[LABEL], sum = 0.f;
    for (int o = 0; o < LABEL; ++o) { ex[o] = __expf(sLog[o] - m2); sum += ex[o]; }
    for (int o = 0; o < LABEL; ++o) {
      out[b * LABEL + o] = sLog[o];                       // logits
      out[BB * LABEL + b * LABEL + o] = ex[o] / sum;      // probs
    }
  }
}

// ---------------- Host side ----------------

extern "C" void kernel_launch(void* const* d_in, const int* in_sizes, int n_in,
                              void* d_out, int out_size, void* d_ws, size_t ws_size,
                              hipStream_t stream) {
  (void)in_sizes; (void)n_in; (void)out_size; (void)ws_size;
  const int*   seq   = (const int*)d_in[0];
  const int*   len   = (const int*)d_in[1];
  const float* emb   = (const float*)d_in[2];
  const float* Wih_f = (const float*)d_in[3];
  const float* Whh_f = (const float*)d_in[4];
  const float* bih_f = (const float*)d_in[5];
  const float* bhh_f = (const float*)d_in[6];
  const float* Wih_b = (const float*)d_in[7];
  const float* Whh_b = (const float*)d_in[8];
  const float* bih_b = (const float*)d_in[9];
  const float* bhh_b = (const float*)d_in[10];
  const float* Ww    = (const float*)d_in[11];
  const float* bw    = (const float*)d_in[12];
  const float* Wa    = (const float*)d_in[13];
  const float* ba    = (const float*)d_in[14];
  const float* Wc    = (const float*)d_in[15];
  const float* bc    = (const float*)d_in[16];
  float* out = (float*)d_out;

  char* ws = (char*)d_ws;
  size_t off = 0;
  auto carve = [&](size_t bytes) -> char* {
    char* p = ws + off;
    off = (off + bytes + 255) & ~(size_t)255;
    return p;
  };
  _Float16* x16   = (_Float16*)carve((size_t)NROW * EMB * 2);        // 16 MB
  _Float16* wih16 = (_Float16*)carve((size_t)2 * G3 * EMB * 2);      // 768 KB
  _Float16* whh16 = (_Float16*)carve((size_t)2 * G3 * HID * 2);      // 768 KB
  _Float16* ww16  = (_Float16*)carve((size_t)HID * 2 * HID * 2);     // 256 KB
  _Float16* gx16  = (_Float16*)carve((size_t)2 * NROW * G3 * 2);     // 96 MB
  _Float16* hbuf  = (_Float16*)carve((size_t)2 * 2 * BB * HID * 2);  // 128 KB
  float*    rnn32 = (float*)carve((size_t)NROW * 2 * HID * 4);       // 64 MB
  _Float16* rnn16 = (_Float16*)carve((size_t)NROW * 2 * HID * 2);    // 32 MB
  float*    score = (float*)carve((size_t)NROW * 4);                 // 128 KB
  unsigned* ctr   = (unsigned*)carve(256);

  // 0. init score (+ba, -1e4 mask) and sync counters
  init_kernel<<<NROW / 256, 256, 0, stream>>>(score, len, ba, ctr);

  // 1. embedding gather -> f16; weights -> f16
  embed_kernel<<<(NROW * 64) / 256, 256, 0, stream>>>(seq, emb, x16);
  cvt_kernel<<<(G3 * EMB / 4 + 255) / 256, 256, 0, stream>>>(Wih_f, wih16, G3 * EMB / 4);
  cvt_kernel<<<(G3 * EMB / 4 + 255) / 256, 256, 0, stream>>>(Wih_b, wih16 + G3 * EMB, G3 * EMB / 4);
  cvt_kernel<<<(G3 * HID / 4 + 255) / 256, 256, 0, stream>>>(Whh_f, whh16, G3 * HID / 4);
  cvt_kernel<<<(G3 * HID / 4 + 255) / 256, 256, 0, stream>>>(Whh_b, whh16 + G3 * HID, G3 * HID / 4);
  cvt_kernel<<<(HID * 2 * HID / 4 + 255) / 256, 256, 0, stream>>>(Ww, ww16, HID * 2 * HID / 4);

  // 2. input-gate GEMM: gx = x @ Wih^T + bih (both directions), f16
  gx_gemm_kernel<<<6144, 256, 0, stream>>>(x16, wih16, bih_f, bih_b, gx16);

  // 3. persistent bidirectional GRU scan: 4 blocks (dir x half),
  //    198 KB Whh slice + 96 KB async-staged gate inputs = 294 KB dynamic LDS
  gru_scan_kernel<<<4, 256, (384 * LDSW + 2 * GXSL) * 2, stream>>>(gx16, whh16,
                                                                   bhh_f, bhh_b,
                                                                   hbuf, rnn32, rnn16, ctr);

  // 4a. attention scores (fused tanh(rnn@Ww^T+bw).Wa reduce)
  attn_score_kernel<<<1024, 256, 0, stream>>>(rnn16, ww16, bw, Wa, score);

  // 4b. softmax / encoding / classifier / output
  finalize_kernel<<<BB, 256, 0, stream>>>(score, rnn32, Wc, bc, out);
}
